// EMCGCN_15710990369231
// MI455X (gfx1250) — matrix operand order repaired and checked by reading, not compiled, not run on hardware
//
#include <hip/hip_runtime.h>
#include <hip/hip_bf16.h>

#define BB   4
#define SS   384
#define DD   300
#define HH   2
#define EE   20
#define HDIM 150
#define MHID 128
#define DEE  10

typedef _Float16 half16 __attribute__((ext_vector_type(16)));
typedef _Float16 half8  __attribute__((ext_vector_type(8)));
typedef float    float8 __attribute__((ext_vector_type(8)));

static __device__ __forceinline__ float8 zero8() {
  float8 z = {0.f, 0.f, 0.f, 0.f, 0.f, 0.f, 0.f, 0.f};
  return z;
}

static __device__ __forceinline__ float8 wmma_f16(half16 a, half16 b, float8 c) {
  // v_wmma_f32_16x16x32_f16 : D = A(16x32 f16) * B(32x16 f16) + C(16x16 f32)
  return __builtin_amdgcn_wmma_f32_16x16x32_f16(false, a, false, b, (short)0, c,
                                                false, false);
}

// ---------------------------------------------------------------------------
// K1: feature = gcn_inputs @ lin_w + lin_b   ([1536,300]x[300,300], fp32)
//     also writes f16 copy laid out [B,H,S,HD] for the aggregation WMMA B-matrix
// ---------------------------------------------------------------------------
__global__ void k_feat(const float* __restrict__ x, const float* __restrict__ w,
                       const float* __restrict__ bias, float* __restrict__ feat,
                       _Float16* __restrict__ feath) {
  const int row = blockIdx.x;  // b*SS + s
  __shared__ float sx[DD];
  for (int t = threadIdx.x; t < DD; t += blockDim.x) sx[t] = x[(size_t)row * DD + t];
  __syncthreads();
  const int t = threadIdx.x;
  if (t < DD) {
    float acc = bias[t];
    for (int k = 0; k < DD; ++k) acc += sx[k] * w[(size_t)k * DD + t];
    feat[(size_t)row * DD + t] = acc;
    const int b = row / SS, s = row % SS, h = t / HDIM, d = t % HDIM;
    feath[((size_t)(b * HH + h) * SS + s) * HDIM + d] = (_Float16)acc;
  }
}

// ---------------------------------------------------------------------------
// K2: attn_src / attn_dst : per (b,h,s) dot of feature head-slice with fc_w
// ---------------------------------------------------------------------------
__global__ void k_score(const float* __restrict__ feat, const float* __restrict__ fw1,
                        const float* __restrict__ fw2, float* __restrict__ asrc,
                        float* __restrict__ adst) {
  const int idx = blockIdx.x * blockDim.x + threadIdx.x;  // (b*HH+h)*SS + s
  if (idx >= BB * HH * SS) return;
  const int s = idx % SS, h = (idx / SS) % HH, b = idx / (HH * SS);
  const float* fr = feat + ((size_t)b * SS + s) * DD + h * HDIM;
  float a1 = 0.f, a2 = 0.f;
  for (int d = 0; d < HDIM; ++d) {
    const float f = fr[d];
    a1 += f * fw1[h * HDIM + d];
    a2 += f * fw2[h * HDIM + d];
  }
  asrc[idx] = a1;
  adst[idx] = a2;
}

// ---------------------------------------------------------------------------
// K3 (WMMA): edge MLP  A = relu(wps @ w1 + b1) @ w2 + b2  -> logits [B,H,S,S]
//            also mask[b,i,j] = (sum_e wps == 0)
// block = 128 thr (4 waves) handles one (b,i); wave handles 16-row j tiles.
//  - wps tile (16 rows x 20 floats, contiguous 1280B) is staged to LDS with
//    coalesced loads, then gathered into the WMMA A layout.
//  - B fragments pre-packed in LDS in per-lane layout: one aligned 32B load/use.
//  - only one accumulator live at a time (no spills).
// ---------------------------------------------------------------------------
__global__ __launch_bounds__(128, 1) void k_edge_mlp(
    const float* __restrict__ wp, const float* __restrict__ w1,
    const float* __restrict__ b1, const float* __restrict__ w2,
    const float* __restrict__ b2, float* __restrict__ logits,
    unsigned char* __restrict__ mask) {
  __shared__ half16 s_b1f[8 * 32];  // 8 fragments x 32 lanes x 16 halves (8 KB)
  __shared__ half16 s_b2f[4 * 32];  // 4 fragments (4 KB)
  __shared__ float s_b1[MHID];
  __shared__ __align__(16) _Float16 s_hid[4][16 * MHID];  // per-wave hidden tile
  __shared__ float s_wp[4][16 * EE];                      // per-wave wps tile

  const int tid = threadIdx.x;
  {
    // Pack GEMM1 B fragments: frag nt, lane L, half t -> K=(L>=16?8:0)+t+(t>=8?8:0),
    // col = nt*16 + (L&15); zero-pad K >= 20.
    _Float16* p = (_Float16*)s_b1f;
    for (int idx = tid; idx < 8 * 32 * 16; idx += 128) {
      const int t = idx & 15;
      const int L = (idx >> 4) & 31;
      const int nt = idx >> 9;
      const int K = ((L & 16) ? 8 : 0) + t + ((t >= 8) ? 8 : 0);
      const int col = nt * 16 + (L & 15);
      p[idx] = (_Float16)((K < EE) ? w1[K * MHID + col] : 0.f);
    }
    // Pack GEMM2 B fragments: frag kc, lane L, half t -> K=kc*32+(L>=16?8:0)+t+(t>=8?8:0),
    // col = L&15 (valid < HH).
    _Float16* q = (_Float16*)s_b2f;
    for (int idx = tid; idx < 4 * 32 * 16; idx += 128) {
      const int t = idx & 15;
      const int L = (idx >> 4) & 31;
      const int kc = idx >> 9;
      const int K = kc * 32 + ((L & 16) ? 8 : 0) + t + ((t >= 8) ? 8 : 0);
      const int col = L & 15;
      q[idx] = (_Float16)((col < HH) ? w2[K * HH + col] : 0.f);
    }
    for (int t = tid; t < MHID; t += 128) s_b1[t] = b1[t];
  }
  __syncthreads();

  const int bid = blockIdx.x;
  const int b = bid / SS, i = bid % SS;
  const int wv = tid >> 5, lane = tid & 31;
  const int lrow = lane & 15;  // fragment row / col selector
  const int hiK = lane >> 4;   // 0: lanes 0-15, 1: lanes 16-31

  for (int mt = wv; mt < SS / 16; mt += 4) {
    const int j0 = mt * 16;
    const int j = j0 + lrow;
    const float diagadd = (i == j) ? 1.f : 0.f;

    // Stage the contiguous 16x20 wps tile: 320 floats, coalesced (10 per lane).
    const float* tbase = wp + (((size_t)b * SS + i) * SS + j0) * EE;
#pragma unroll
    for (int q = 0; q < 10; ++q) s_wp[wv][lane + q * 32] = tbase[lane + q * 32];

    // A fragment: rows = j, K = padded edge dim (20 -> 32); gather from LDS.
    half16 afrag;
    float psum = 0.f;
#pragma unroll
    for (int t = 0; t < 16; ++t) {
      const int K = t + ((t >= 8) ? 8 : 0) + (hiK ? 8 : 0);
      float v = 0.f;
      if (K < EE) { v = s_wp[wv][lrow * EE + K] + diagadd; psum += v; }
      afrag[t] = (_Float16)v;
    }
    // row-sum for mask: lane L covers one half of K, lane L^16 the other
    const float rowsum = psum + __shfl_xor(psum, 16, 32);
    if (hiK == 0) mask[((size_t)b * SS + i) * SS + j] = (rowsum == 0.f) ? 1 : 0;

    // GEMM1: hidden(16x128) = A(16x32) @ w1(32x128); consume each n-tile at once
#pragma unroll
    for (int nt = 0; nt < 8; ++nt) {
      const half16 bf = s_b1f[nt * 32 + lane];
      const float8 acc = wmma_f16(afrag, bf, zero8());
      const int hcol = nt * 16 + lrow;
      const float bias = s_b1[hcol];
#pragma unroll
      for (int v = 0; v < 8; ++v) {
        const int r = v + hiK * 8;
        float hv = acc[v] + bias;
        hv = fmaxf(hv, 0.f);
        s_hid[wv][r * MHID + hcol] = (_Float16)hv;
      }
    }

    // GEMM2: out(16x16, cols 0..1 valid) = hidden(16x128) @ w2(128x2 pad 16)
    float8 acc2 = zero8();
#pragma unroll
    for (int kc = 0; kc < 4; ++kc) {
      // A' halves: t<8 -> K=kc*32+hiK*8+t ; t>=8 -> K=kc*32+hiK*8+16+(t-8)
      const int base = lrow * MHID + kc * 32 + hiK * 8;
      const half8 lo = *(const half8*)&s_hid[wv][base];
      const half8 hi = *(const half8*)&s_hid[wv][base + 16];
      half16 a2;
#pragma unroll
      for (int t = 0; t < 8; ++t) { a2[t] = lo[t]; a2[t + 8] = hi[t]; }
      acc2 = wmma_f16(a2, s_b2f[kc * 32 + lane], acc2);
    }

    if (lrow < HH) {
      const float bias2 = b2[lrow];
      // 8 consecutive j outputs -> two 16B stores
      float* lbase =
          logits + ((((size_t)b * HH + lrow) * SS + i) * SS + j0 + hiK * 8);
      float4 o0 = make_float4(acc2[0] + bias2, acc2[1] + bias2, acc2[2] + bias2,
                              acc2[3] + bias2);
      float4 o1 = make_float4(acc2[4] + bias2, acc2[5] + bias2, acc2[6] + bias2,
                              acc2[7] + bias2);
      *(float4*)lbase = o0;
      *((float4*)lbase + 1) = o1;
    }
  }
}

// ---------------------------------------------------------------------------
// K4: softmax over j of leaky_relu(A + src_i + dst_j) with mask -> attn (f16)
// one block (128 thr) per (b,h,i) row of 384
// ---------------------------------------------------------------------------
__global__ void k_softmax(const float* __restrict__ logits, const float* __restrict__ asrc,
                          const float* __restrict__ adst, const unsigned char* __restrict__ mask,
                          _Float16* __restrict__ attn) {
  const int r = blockIdx.x;  // (b*HH+h)*SS + i
  const int i = r % SS, h = (r / SS) % HH, b = r / (HH * SS);
  const float* lrow = logits + (size_t)r * SS;
  const float srcv = asrc[(b * HH + h) * SS + i];
  const float* drow = adst + (size_t)(b * HH + h) * SS;
  const unsigned char* mrow = mask + ((size_t)b * SS + i) * SS;

  __shared__ float red[128];
  const int t = threadIdx.x;
  float vals[3];
  float mymax = -INFINITY;
#pragma unroll
  for (int q = 0; q < 3; ++q) {
    const int j = t + q * 128;
    float x = lrow[j] + srcv + drow[j];
    x = (x >= 0.f) ? x : 0.01f * x;  // leaky_relu(0.01)
    if (mrow[j]) x = -INFINITY;
    vals[q] = x;
    mymax = fmaxf(mymax, x);
  }
  red[t] = mymax;
  __syncthreads();
  for (int o = 64; o > 0; o >>= 1) {
    if (t < o) red[t] = fmaxf(red[t], red[t + o]);
    __syncthreads();
  }
  const float m = red[0];
  __syncthreads();
  float s = 0.f;
#pragma unroll
  for (int q = 0; q < 3; ++q) {
    const float e = __expf(vals[q] - m);
    vals[q] = e;
    s += e;
  }
  red[t] = s;
  __syncthreads();
  for (int o = 64; o > 0; o >>= 1) {
    if (t < o) red[t] += red[t + o];
    __syncthreads();
  }
  const float inv = 1.f / red[0];
#pragma unroll
  for (int q = 0; q < 3; ++q)
    attn[(size_t)r * SS + t + q * 128] = (_Float16)(vals[q] * inv);
}

// ---------------------------------------------------------------------------
// K5 (WMMA): aggregation  gcn_mid[b,i,h*HD+d] = sum_j attn[b,h,i,j]*feat[b,h,j,d]
// per (b,h): 384x384x150 GEMM, tiles 16x16x32; one wave per (mtile,ntile)
// ---------------------------------------------------------------------------
__global__ __launch_bounds__(128, 1) void k_agg(const _Float16* __restrict__ attn,
                                                const _Float16* __restrict__ feath,
                                                float* __restrict__ gcnmid) {
  const int lane = threadIdx.x & 31;
  const int wv = threadIdx.x >> 5;
  const int gw = blockIdx.x * 4 + wv;  // 0 .. BB*HH*24*10-1 (grid sized exactly)
  const int NT = 10, MT = 24;
  const int nt = gw % NT;
  const int mt = (gw / NT) % MT;
  const int h = (gw / (NT * MT)) % HH;
  const int b = gw / (NT * MT * HH);

  const int lrow = lane & 15;
  const int hiK = lane >> 4;
  const int i0 = mt * 16, d0 = nt * 16;
  const int d = d0 + lrow;

  const _Float16* aptr = attn + (size_t)(b * HH + h) * SS * SS;
  const _Float16* fptr = feath + (size_t)(b * HH + h) * SS * HDIM;

  float8 acc = zero8();
  for (int kt = 0; kt < SS / 32; ++kt) {
    half16 af;
#pragma unroll
    for (int t = 0; t < 16; ++t) {
      const int j = kt * 32 + t + ((t >= 8) ? 8 : 0) + hiK * 8;
      af[t] = aptr[(size_t)(i0 + lrow) * SS + j];
    }
    half16 bf;
#pragma unroll
    for (int t = 0; t < 16; ++t) {
      const int j = kt * 32 + t + ((t >= 8) ? 8 : 0) + hiK * 8;
      bf[t] = (d < HDIM) ? fptr[(size_t)j * HDIM + d] : (_Float16)0.f;
    }
    acc = wmma_f16(af, bf, acc);
  }
  if (d < HDIM) {
#pragma unroll
    for (int v = 0; v < 8; ++v) {
      const int i = i0 + v + hiK * 8;
      gcnmid[((size_t)b * SS + i) * DD + h * HDIM + d] = acc[v];
    }
  }
}

// ---------------------------------------------------------------------------
// K6: gcn_out = gcn_mid @ W_w + W_b ; torch LayerNorm (ddof=1, eps on std);
//     node = relu(.)  -> written straight to d_out
// ---------------------------------------------------------------------------
__global__ void k_wln(const float* __restrict__ gcnmid, const float* __restrict__ Ww,
                      const float* __restrict__ Wb, const float* __restrict__ lna,
                      const float* __restrict__ lnb, float* __restrict__ node) {
  const int row = blockIdx.x;
  __shared__ float sx[DD];
  __shared__ float red[512];
  __shared__ float sstat[2];
  const int t = threadIdx.x;
  for (int k = t; k < DD; k += blockDim.x) sx[k] = gcnmid[(size_t)row * DD + k];
  __syncthreads();
  float acc = 0.f;
  if (t < DD) {
    acc = Wb[t];
    for (int k = 0; k < DD; ++k) acc += sx[k] * Ww[(size_t)k * DD + t];
  }
  red[t] = (t < DD) ? acc : 0.f;
  __syncthreads();
  for (int o = 256; o > 0; o >>= 1) {
    if (t < o) red[t] += red[t + o];
    __syncthreads();
  }
  if (t == 0) sstat[0] = red[0] * (1.f / (float)DD);
  __syncthreads();
  const float mean = sstat[0];
  const float dv = (t < DD) ? (acc - mean) : 0.f;
  red[t] = dv * dv;
  __syncthreads();
  for (int o = 256; o > 0; o >>= 1) {
    if (t < o) red[t] += red[t + o];
    __syncthreads();
  }
  if (t == 0) sstat[1] = sqrtf(red[0] * (1.f / (float)(DD - 1)));
  __syncthreads();
  if (t < DD) {
    const float y = lna[t] * (acc - mean) / (sstat[1] + 1e-6f) + lnb[t];
    node[(size_t)row * DD + t] = fmaxf(y, 0.f);
  }
}

// ---------------------------------------------------------------------------
// K7: diag[b,s,e] = weight_adj[b,s,s,e]
// ---------------------------------------------------------------------------
__global__ void k_diag(const float* __restrict__ wadj, float* __restrict__ diag) {
  const int idx = blockIdx.x * blockDim.x + threadIdx.x;
  if (idx >= BB * SS * EE) return;
  const int e = idx % EE;
  const int row = idx / EE;  // b*SS + s
  const int b = row / SS, s = row % SS;
  diag[idx] = wadj[(((size_t)b * SS + s) * SS + s) * EE + e];
}

// ---------------------------------------------------------------------------
// K8: per-node highway precompute.
//  Qj[b,s,o] = diag[b,s]@hw[20:40] + node[b,s]@hw[60:360]          (j-dependent)
//  Qi[b,s,o] = diag[b,s]@hw[40:60] + node[b,s]@hw[360:660] + hw_b  (i-dependent)
// ---------------------------------------------------------------------------
__global__ void k_q(const float* __restrict__ diag, const float* __restrict__ node,
                    const float* __restrict__ hw, const float* __restrict__ hb,
                    float* __restrict__ Qi, float* __restrict__ Qj) {
  const int row = blockIdx.x;  // b*SS + s
  const int lane = threadIdx.x;
  const float* dg = diag + (size_t)row * EE;
  const float* nd = node + (size_t)row * DD;
  if (lane < DEE) {
    float q = 0.f;
    for (int e = 0; e < EE; ++e) q += dg[e] * hw[(EE + e) * DEE + lane];
    for (int d = 0; d < DD; ++d) q += nd[d] * hw[(3 * EE + d) * DEE + lane];
    Qj[(size_t)row * DEE + lane] = q;
  } else if (lane >= 16 && lane < 16 + DEE) {
    const int o = lane - 16;
    float q = hb[o];
    for (int e = 0; e < EE; ++e) q += dg[e] * hw[(2 * EE + e) * DEE + o];
    for (int d = 0; d < DD; ++d) q += nd[d] * hw[(3 * EE + DD + d) * DEE + o];
    Qi[(size_t)row * DEE + o] = q;
  }
}

// ---------------------------------------------------------------------------
// K9: edge_out[b,i,j,o] = wadj[b,i,j,:]@hw[0:20,o] + Qj[b,j,o] + Qi[b,i,o]
// ---------------------------------------------------------------------------
__global__ void k_edgeout(const float* __restrict__ wadj, const float* __restrict__ hw,
                          const float* __restrict__ Qi, const float* __restrict__ Qj,
                          float* __restrict__ eout) {
  __shared__ float s_h0[EE * DEE];
  for (int t = threadIdx.x; t < EE * DEE; t += blockDim.x) s_h0[t] = hw[t];
  __syncthreads();
  const size_t p = (size_t)blockIdx.x * blockDim.x + threadIdx.x;
  if (p >= (size_t)BB * SS * SS) return;
  const int j = (int)(p % SS);
  const int i = (int)((p / SS) % SS);
  const int b = (int)(p / (SS * SS));
  const float* wr = wadj + p * EE;
  float a[EE];
#pragma unroll
  for (int e = 0; e < EE; ++e) a[e] = wr[e];
  const float* qj = Qj + ((size_t)b * SS + j) * DEE;
  const float* qi = Qi + ((size_t)b * SS + i) * DEE;
#pragma unroll
  for (int o = 0; o < DEE; ++o) {
    float acc = qi[o] + qj[o];
#pragma unroll
    for (int e = 0; e < EE; ++e) acc += a[e] * s_h0[e * DEE + o];
    eout[p * DEE + o] = acc;
  }
}

// ---------------------------------------------------------------------------
extern "C" void kernel_launch(void* const* d_in, const int* in_sizes, int n_in,
                              void* d_out, int out_size, void* d_ws, size_t ws_size,
                              hipStream_t stream) {
  const float* wp     = (const float*)d_in[0];
  const float* wadj   = (const float*)d_in[1];
  const float* gin    = (const float*)d_in[2];
  // d_in[3] self_loop: identity over (i,j) expanded over E — reconstructed analytically
  const float* lin_w  = (const float*)d_in[4];
  const float* lin_b  = (const float*)d_in[5];
  const float* fc_w1  = (const float*)d_in[6];
  const float* fc_w2  = (const float*)d_in[7];
  const float* mlp_w1 = (const float*)d_in[8];
  const float* mlp_b1 = (const float*)d_in[9];
  const float* mlp_w2 = (const float*)d_in[10];
  const float* mlp_b2 = (const float*)d_in[11];
  const float* W_w    = (const float*)d_in[12];
  const float* W_b    = (const float*)d_in[13];
  const float* ln_a   = (const float*)d_in[14];
  const float* ln_b   = (const float*)d_in[15];
  const float* hw_w   = (const float*)d_in[16];
  const float* hw_b   = (const float*)d_in[17];

  char* ws = (char*)d_ws;
  float*         feat   = (float*)(ws + 0);           //  460800 f
  float*         gcnmid = (float*)(ws + 1843200);     //  460800 f
  float*         logits = (float*)(ws + 3686400);     // 1179648 f
  float*         asrc   = (float*)(ws + 8404992);     //    3072 f
  float*         adst   = (float*)(ws + 8417280);     //    3072 f
  float*         diag   = (float*)(ws + 8429568);     //   30720 f
  float*         Qi     = (float*)(ws + 8552448);     //   15360 f
  float*         Qj     = (float*)(ws + 8613888);     //   15360 f
  _Float16*      feath  = (_Float16*)(ws + 8675328);  //  460800 h
  _Float16*      attn   = (_Float16*)(ws + 9596928);  // 1179648 h
  unsigned char* mask   = (unsigned char*)(ws + 11956224);  // 589824 B

  float* node = (float*)d_out;                 // [B,S,D]
  float* eout = (float*)d_out + BB * SS * DD;  // [B,S,S,DE]

  k_feat<<<BB * SS, 320, 0, stream>>>(gin, lin_w, lin_b, feat, feath);
  k_score<<<(BB * HH * SS + 255) / 256, 256, 0, stream>>>(feat, fc_w1, fc_w2, asrc, adst);
  k_edge_mlp<<<BB * SS, 128, 0, stream>>>(wp, mlp_w1, mlp_b1, mlp_w2, mlp_b2, logits, mask);
  k_softmax<<<BB * HH * SS, 128, 0, stream>>>(logits, asrc, adst, mask, attn);
  k_agg<<<BB * HH * 24 * 10 / 4, 128, 0, stream>>>(attn, feath, gcnmid);
  k_wln<<<BB * SS, 512, 0, stream>>>(gcnmid, W_w, W_b, ln_a, ln_b, node);
  k_diag<<<(BB * SS * EE + 255) / 256, 256, 0, stream>>>(wadj, diag);
  k_q<<<BB * SS, 32, 0, stream>>>(diag, node, hw_w, hw_b, Qi, Qj);
  k_edgeout<<<(BB * SS * SS) / 256, 256, 0, stream>>>(wadj, hw_w, Qi, Qj, eout);
}
